// RelativeDotProductAttention_10642928959689
// MI455X (gfx1250) — matrix operand order repaired and checked by reading, not compiled, not run on hardware
//
#include <hip/hip_runtime.h>
#include <hip/hip_bf16.h>
#include <math.h>

// ---------------------------------------------------------------------------
// Relative dot-product attention (Shaw-style), B=8 H=8 S=1024 D=64, causal.
//   out[0] : output  (B,H,S,D) f32     out[1] : weights (B,H,S,S) f32
//
// Prep    : Q,K,Wk -> bf16 (fragment-ready, row-major); V,Wv -> bf16
//           TRANSPOSED so kernel-2 B-fragments are contiguous b128 loads.
// Kernel 1: logits = QK^T + band-WMMA rel-K (ds_bpermute skew) -> softmax in
//           LDS (vectorized b128 scans) -> write weights (zeros above diag).
// Kernel 2: output = W @ V + band-WMMA rel-V (skewed A-frag of W).
// All matrix math: v_wmma_f32_16x16x32_bf16 (wave32).  Falls back to direct
// f32 fragment builds if the workspace is too small (PRE=false).
// ---------------------------------------------------------------------------

typedef __attribute__((ext_vector_type(16))) __bf16       bf16x16;
typedef __attribute__((ext_vector_type(8)))  __bf16       bf16x8;
typedef __attribute__((ext_vector_type(8)))  float        f32x8;
typedef __attribute__((ext_vector_type(4)))  unsigned int u32x4;

static __device__ __forceinline__ f32x8 wmma_bf16(bf16x16 a, bf16x16 b, f32x8 c) {
  return __builtin_amdgcn_wmma_f32_16x16x32_bf16(false, a, false, b, (short)0, c,
                                                 false, false);
}

static __device__ __forceinline__ float bpermute_f32(int src_lane, float v) {
  int r = __builtin_amdgcn_ds_bpermute(src_lane << 2, __builtin_bit_cast(int, v));
  return __builtin_bit_cast(float, r);
}

// B-fragment: 16 contiguous bf16 (32B, 16B-aligned) -> two b128 loads.
static __device__ __forceinline__ bf16x16 ld_b_frag(const __bf16* p) {
  union { u32x4 q[2]; bf16x16 v; } u;
  u.q[0] = *(const u32x4*)(p);
  u.q[1] = *(const u32x4*)(p + 8);
  return u.v;
}
// A-fragment: two 8-element bf16 runs at p and p+16 (each 16B-aligned).
static __device__ __forceinline__ bf16x16 ld_a_frag(const __bf16* p) {
  union { u32x4 q[2]; bf16x16 v; } u;
  u.q[0] = *(const u32x4*)(p);
  u.q[1] = *(const u32x4*)(p + 16);
  return u.v;
}

#define SEQ        1024
#define HEADDIM    64
#define MAXREL     512
#define VOCAB      1025
#define NBH        64
#define WVT_STRIDE 1040
#define OUT_ELEMS  ((size_t)NBH * SEQ * HEADDIM)        // 4,194,304

// workspace layout (bytes)
#define QKV_EL     ((size_t)NBH * SEQ * HEADDIM)
#define OFF_Q      ((size_t)0)
#define OFF_K      ((size_t)8388608)
#define OFF_VT     ((size_t)16777216)
#define OFF_WK     ((size_t)25165824)
#define OFF_WVT    ((size_t)25165824 + 131328)
#define WS_NEEDED  ((size_t)25165824 + 131328 + 133120)

static __device__ __forceinline__ int clip_rel(int off) {
  return (off < -MAXREL ? -MAXREL : (off > MAXREL ? MAXREL : off)) + MAXREL;
}

// ---------------------------------------------------------------------------
// Prep kernels
// ---------------------------------------------------------------------------
__global__ __launch_bounds__(256) void conv_f32_bf16(
    const float* __restrict__ s, __bf16* __restrict__ d, int n4) {
  int i = blockIdx.x * blockDim.x + threadIdx.x;
  if (i >= n4) return;
  const float4 f = ((const float4*)s)[i];
  union { __bf16 h[4]; unsigned long long u; } u;
  u.h[0] = (__bf16)f.x; u.h[1] = (__bf16)f.y;
  u.h[2] = (__bf16)f.z; u.h[3] = (__bf16)f.w;
  ((unsigned long long*)d)[i] = u.u;
}

// Vt[bh][d][j] = (bf16) V[bh][j][d] ; coalesced reads over d.
__global__ __launch_bounds__(256) void transpose_v_bf16(
    const float* __restrict__ v, __bf16* __restrict__ vt) {
  const int bh   = blockIdx.x >> 4;
  const int part = blockIdx.x & 15;              // 64-column slab of j
  const int d    = threadIdx.x & 63;
  const int jq   = threadIdx.x >> 6;             // 0..3
  const float* Vb  = v  + (size_t)bh * SEQ * HEADDIM;
  __bf16*      Vtb = vt + ((size_t)bh * HEADDIM + d) * SEQ;
  const int j0 = part * 64 + jq * 16;
  union { __bf16 h[16]; u32x4 q[2]; } u;
#pragma unroll
  for (int e = 0; e < 16; ++e)
    u.h[e] = (__bf16)Vb[(size_t)(j0 + e) * HEADDIM + d];
  *(u32x4*)(Vtb + j0)     = u.q[0];
  *(u32x4*)(Vtb + j0 + 8) = u.q[1];
}

// Wvt[d][p] = (bf16) Wv[p][d]
__global__ __launch_bounds__(256) void transpose_wv_bf16(
    const float* __restrict__ wv, __bf16* __restrict__ wvt) {
  int i = blockIdx.x * blockDim.x + threadIdx.x;
  if (i >= HEADDIM * VOCAB) return;
  int d = i / VOCAB, p = i - d * VOCAB;
  wvt[(size_t)d * WVT_STRIDE + p] = (__bf16)wv[(size_t)p * HEADDIM + d];
}

// ---------------------------------------------------------------------------
// Kernel 1: one wave per (bh, 16-row tile). grid = 64*64 = 4096 blocks.
// ---------------------------------------------------------------------------
template <bool PRE>
__global__ __launch_bounds__(32) void rel_attn_logits_softmax(
    const void* __restrict__ qv, const void* __restrict__ kv,
    const void* __restrict__ wkv, float* __restrict__ out_w) {
  __shared__ __bf16 P[16][SEQ + 8];            // 16 logit rows, bf16, padded

  const int l    = threadIdx.x;
  const int bh   = blockIdx.x >> 6;
  const int i0   = (blockIdx.x & 63) << 4;
  const int half = l >> 4;
  const int m    = l & 15;
  const int kb   = half * 16;

  // ---- A fragments of Q (row i0+m), K-halves d=0..31 / 32..63 ----
  bf16x16 aq0, aq1;
  if constexpr (PRE) {
    const __bf16* qr = (const __bf16*)qv + ((size_t)bh * SEQ + i0 + m) * HEADDIM;
    aq0 = ld_b_frag(qr + kb);
    aq1 = ld_b_frag(qr + 32 + kb);
  } else {
    const float* qr = (const float*)qv + ((size_t)bh * SEQ + i0 + m) * HEADDIM;
#pragma unroll
    for (int e = 0; e < 8; ++e) {
      aq0[e]     = (__bf16)qr[kb + e];
      aq0[8 + e] = (__bf16)qr[kb + 8 + e];
      aq1[e]     = (__bf16)qr[32 + kb + e];
      aq1[8 + e] = (__bf16)qr[32 + kb + 8 + e];
    }
  }

  const int ntiles = (i0 >> 4) + 1;
  for (int jt = 0; jt < ntiles; ++jt) {
    const int j0 = jt << 4;

    // ---- QK^T B-fragments: column n=m is key row j0+m; K-dim is d ----
    bf16x16 bk0, bk1;
    if constexpr (PRE) {
      const __bf16* kr =
          (const __bf16*)kv + ((size_t)bh * SEQ + j0 + m) * HEADDIM + half * 16;
      bk0 = ld_b_frag(kr);
      bk1 = ld_b_frag(kr + 32);
      if (jt + 1 < ntiles) __builtin_prefetch(kr + 16 * HEADDIM, 0, 1);
    } else {
      const float* kr =
          (const float*)kv + ((size_t)bh * SEQ + j0 + m) * HEADDIM + half * 16;
#pragma unroll
      for (int e = 0; e < 16; ++e) {
        bk0[e] = (__bf16)kr[e];
        bk1[e] = (__bf16)kr[32 + e];
      }
    }
    f32x8 c = {};
    c = wmma_bf16(aq0, bk0, c);
    c = wmma_bf16(aq1, bk1, c);

    // ---- rel-key band: R[ii,t] = q_{i0+ii} . Wk[clip(base+t-15)] ----
    const int base = j0 - i0;
    f32x8 r[2];
#pragma unroll
    for (int g = 0; g < 2; ++g) {
      const int t = g * 16 + m;
      const size_t woff = (size_t)clip_rel(base + t - 15) * HEADDIM + half * 16;
      bf16x16 b0, b1;
      if constexpr (PRE) {
        const __bf16* wr = (const __bf16*)wkv + woff;
        b0 = ld_b_frag(wr);
        b1 = ld_b_frag(wr + 32);
      } else {
        const float* wr = (const float*)wkv + woff;
#pragma unroll
        for (int e = 0; e < 16; ++e) {
          b0[e] = (__bf16)wr[e];
          b1[e] = (__bf16)wr[32 + e];
        }
      }
      f32x8 rr = {};
      rr = wmma_bf16(aq0, b0, rr);
      rr = wmma_bf16(aq1, b1, rr);
      r[g] = rr;
    }

    // ---- skew gather: c[ii][jj] += R[ii][t=jj-ii+15] via ds_bpermute ----
#pragma unroll
    for (int rr = 0; rr < 8; ++rr) {
      const int ii = rr + half * 8;
      const int t  = m - ii + 15;                 // in [0,30]
      const int sl = (t & 15) + (half << 4);      // same half, same VGPR
      const float g0 = bpermute_f32(sl, r[0][rr]);
      const float g1 = bpermute_f32(sl, r[1][rr]);
      c[rr] += (t < 16) ? g0 : g1;
    }

    if (j0 == i0) {                               // causal mask, diagonal tile
#pragma unroll
      for (int rr = 0; rr < 8; ++rr) {
        const int ii = rr + half * 8;
        if (m > ii) c[rr] = -INFINITY;
      }
    }
#pragma unroll
    for (int rr = 0; rr < 8; ++rr) P[rr + half * 8][j0 + m] = (__bf16)c[rr];
  }
  __syncthreads();

  // ---- per-row softmax over valid columns [0, i]; 2 lanes per row ----
  const int rrow = l >> 1, part = l & 1;
  const int i  = i0 + rrow;
  const int n  = i + 1;
  const int js = part * 512;
  int je = js + 512; if (je > n) je = n; if (je < js) je = js;
  const int jev = js + ((je - js) & ~7);

  const bf16x8* rvc = (const bf16x8*)&P[rrow][0];
  float mx = -INFINITY;
  for (int j = js; j < jev; j += 8) {
    bf16x8 x = rvc[j >> 3];
#pragma unroll
    for (int e = 0; e < 8; ++e) mx = fmaxf(mx, (float)x[e]);
  }
  for (int j = jev; j < je; ++j) mx = fmaxf(mx, (float)P[rrow][j]);
  mx = fmaxf(mx, __shfl_xor(mx, 1));

  float s = 0.f;
  bf16x8* rvm = (bf16x8*)&P[rrow][0];
  for (int j = js; j < jev; j += 8) {
    bf16x8 x = rvm[j >> 3], y;
#pragma unroll
    for (int e = 0; e < 8; ++e) {
      const float ev = __expf((float)x[e] - mx);
      s += ev;
      y[e] = (__bf16)ev;
    }
    rvm[j >> 3] = y;
  }
  for (int j = jev; j < je; ++j) {
    const float ev = __expf((float)P[rrow][j] - mx);
    s += ev;
    P[rrow][j] = (__bf16)ev;
  }
  s += __shfl_xor(s, 1);
  const float inv = 1.f / s;
  __syncthreads();

  // ---- write full weight rows (normalized; zeros above diagonal) ----
  float* wout = out_w + ((size_t)bh * SEQ + i) * SEQ;
  for (int j = js; j < js + 512; j += 8) {
    bf16x8 x = rvc[j >> 3];
    float4 o0, o1;
    o0.x = (j + 0 < n) ? (float)x[0] * inv : 0.f;
    o0.y = (j + 1 < n) ? (float)x[1] * inv : 0.f;
    o0.z = (j + 2 < n) ? (float)x[2] * inv : 0.f;
    o0.w = (j + 3 < n) ? (float)x[3] * inv : 0.f;
    o1.x = (j + 4 < n) ? (float)x[4] * inv : 0.f;
    o1.y = (j + 5 < n) ? (float)x[5] * inv : 0.f;
    o1.z = (j + 6 < n) ? (float)x[6] * inv : 0.f;
    o1.w = (j + 7 < n) ? (float)x[7] * inv : 0.f;
    *(float4*)(wout + j)     = o0;
    *(float4*)(wout + j + 4) = o1;
  }
}

// ---------------------------------------------------------------------------
// Kernel 2: output = W @ V + skew(W) @ WvBand. One wave per 16-row tile.
// ---------------------------------------------------------------------------
template <bool PRE>
__global__ __launch_bounds__(32) void rel_attn_output(
    const void* __restrict__ vv, const void* __restrict__ wvv,
    const float* __restrict__ w, float* __restrict__ out) {
  const int l    = threadIdx.x;
  const int bh   = blockIdx.x >> 6;
  const int i0   = (blockIdx.x & 63) << 4;
  const int half = l >> 4;
  const int m    = l & 15;
  const int kb   = half * 16;

  const float* wrow = w + ((size_t)bh * SEQ + i0 + m) * SEQ;

  f32x8 acc[4] = {{}, {}, {}, {}};

  // ---- dense P @ V, K = j in chunks of 32 ----
  for (int j0 = 0; j0 < i0 + 16; j0 += 32) {
    bf16x16 ap;
#pragma unroll
    for (int e = 0; e < 8; ++e) {
      ap[e]     = (__bf16)wrow[j0 + kb + e];
      ap[8 + e] = (__bf16)wrow[j0 + kb + 8 + e];
    }
    if (j0 + 32 < i0 + 16) __builtin_prefetch(wrow + j0 + 32, 0, 1);
#pragma unroll
    for (int nb = 0; nb < 4; ++nb) {
      bf16x16 bv;
      if constexpr (PRE) {
        const __bf16* vt = (const __bf16*)vv +
            ((size_t)bh * HEADDIM + nb * 16 + m) * SEQ + j0 + half * 16;
        bv = ld_b_frag(vt);
      } else {
        const float* V = (const float*)vv + (size_t)bh * SEQ * HEADDIM;
#pragma unroll
        for (int e = 0; e < 16; ++e)
          bv[e] =
              (__bf16)V[(size_t)(j0 + half * 16 + e) * HEADDIM + nb * 16 + m];
      }
      acc[nb] = wmma_bf16(ap, bv, acc[nb]);
    }
  }

  // ---- rel-value band per 16-col tile: skew(W) (16x32) @ WvBand ----
  const int ntiles = (i0 >> 4) + 1;
  for (int jt = 0; jt < ntiles; ++jt) {
    const int j0   = jt << 4;
    const int base = j0 - i0;

    bf16x16 as;                                   // S[ii,t] = w[ii, ii+t-15]
#pragma unroll
    for (int e = 0; e < 8; ++e) {
      const int jj0 = m + (kb + e) - 15, jj1 = m + (kb + 8 + e) - 15;
      as[e]     = (jj0 >= 0 && jj0 < 16) ? (__bf16)wrow[j0 + jj0] : (__bf16)0.f;
      as[8 + e] = (jj1 >= 0 && jj1 < 16) ? (__bf16)wrow[j0 + jj1] : (__bf16)0.f;
    }
#pragma unroll
    for (int nb = 0; nb < 4; ++nb) {
      bf16x16 bb;
#pragma unroll
      for (int e = 0; e < 16; ++e) {
        const int p = clip_rel(base + (half * 16 + e) - 15);
        if constexpr (PRE) {
          bb[e] = ((const __bf16*)wvv)[(size_t)(nb * 16 + m) * WVT_STRIDE + p];
        } else {
          bb[e] =
              (__bf16)((const float*)wvv)[(size_t)p * HEADDIM + nb * 16 + m];
        }
      }
      acc[nb] = wmma_bf16(as, bb, acc[nb]);
    }
  }

  // ---- store output tile ----
#pragma unroll
  for (int nb = 0; nb < 4; ++nb)
#pragma unroll
    for (int rr = 0; rr < 8; ++rr) {
      const int ii = rr + half * 8;
      out[((size_t)bh * SEQ + i0 + ii) * HEADDIM + nb * 16 + m] = acc[nb][rr];
    }
}

// ---------------------------------------------------------------------------
extern "C" void kernel_launch(void* const* d_in, const int* in_sizes, int n_in,
                              void* d_out, int out_size, void* d_ws,
                              size_t ws_size, hipStream_t stream) {
  (void)in_sizes; (void)n_in; (void)out_size;
  const float* q  = (const float*)d_in[0];
  const float* k  = (const float*)d_in[1];
  const float* v  = (const float*)d_in[2];
  const float* wk = (const float*)d_in[3];
  const float* wv = (const float*)d_in[4];
  // d_in[5] (rel_positions) and d_in[6] (mask) are reproduced analytically.

  float* out   = (float*)d_out;              // (B,H,S,D)
  float* out_w = out + OUT_ELEMS;            // (B,H,S,S)

  const dim3 grid(NBH * (SEQ / 16));         // 4096 blocks, 1 wave each
  const dim3 block(32);

  if (ws_size >= WS_NEEDED && d_ws != nullptr) {
    char* ws = (char*)d_ws;
    __bf16* qb  = (__bf16*)(ws + OFF_Q);
    __bf16* kb  = (__bf16*)(ws + OFF_K);
    __bf16* vt  = (__bf16*)(ws + OFF_VT);
    __bf16* wkb = (__bf16*)(ws + OFF_WK);
    __bf16* wvt = (__bf16*)(ws + OFF_WVT);

    const int n4  = (int)(QKV_EL / 4);                         // 1,048,576
    const int wk4 = (VOCAB * HEADDIM) / 4;                     // 16,400
    conv_f32_bf16<<<(n4 + 255) / 256, 256, 0, stream>>>(q, qb, n4);
    conv_f32_bf16<<<(n4 + 255) / 256, 256, 0, stream>>>(k, kb, n4);
    conv_f32_bf16<<<(wk4 + 255) / 256, 256, 0, stream>>>(wk, wkb, wk4);
    transpose_v_bf16<<<NBH * 16, 256, 0, stream>>>(v, vt);
    transpose_wv_bf16<<<(HEADDIM * VOCAB + 255) / 256, 256, 0, stream>>>(wv, wvt);

    rel_attn_logits_softmax<true><<<grid, block, 0, stream>>>(qb, kb, wkb, out_w);
    rel_attn_output<true><<<grid, block, 0, stream>>>(vt, wvt, out_w, out);
  } else {
    rel_attn_logits_softmax<false><<<grid, block, 0, stream>>>(q, k, wk, out_w);
    rel_attn_output<false><<<grid, block, 0, stream>>>(v, wv, out_w, out);
  }
}
